// TSA_71768903517120
// MI455X (gfx1250) — matrix operand order repaired and checked by reading, not compiled, not run on hardware
//
#include <hip/hip_runtime.h>
#include <hip/hip_bf16.h>

typedef __bf16 bf16;
typedef __attribute__((ext_vector_type(16))) __bf16 v16bf;
typedef __attribute__((ext_vector_type(8)))  __bf16 v8bf;
typedef __attribute__((ext_vector_type(8)))  float  v8f;

// ---------------------------------------------------------------------------
// fp32 -> bf16 elementwise convert
// ---------------------------------------------------------------------------
__global__ void __launch_bounds__(256) k_f2bf(const float* __restrict__ src,
                                              bf16* __restrict__ dst, size_t n) {
    size_t i = (size_t)blockIdx.x * 256 + threadIdx.x;
    if (i < n) dst[i] = (bf16)src[i];
}

__global__ void __launch_bounds__(256) k_zero_bf16(bf16* p, int n) {
    int i = blockIdx.x * 256 + threadIdx.x;
    if (i < n) p[i] = (bf16)0.0f;
}

// ---------------------------------------------------------------------------
// Pack fp32 weight W[K,N] into per-lane WMMA B-fragment layout (bf16):
//   Bp[(kt*nCt + ct)*512 + lane*16 + i] = W[kt*32 + 8*(lane>>4) + i + (i>=8 ? 8 : 0), ct*16 + (lane&15)]
// so the GEMM loads one contiguous 32B v16bf per lane per (ktile,ctile).
// ---------------------------------------------------------------------------
__global__ void __launch_bounds__(256) k_pack_w(const float* __restrict__ W,
                                                bf16* __restrict__ Bp,
                                                int K, int N) {
    size_t idx = (size_t)blockIdx.x * 256 + threadIdx.x;
    size_t total = (size_t)K * N;
    if (idx >= total) return;
    int i    = (int)(idx & 15);
    int lane = (int)((idx >> 4) & 31);
    size_t frag = idx >> 9;          // kt*nCt + ct
    int nCt = N >> 4;
    int ct = (int)(frag % nCt);
    int kt = (int)(frag / nCt);
    int kk = kt * 32 + ((lane >> 4) << 3) + i + ((i >> 3) << 3);
    int cc = (ct << 4) + (lane & 15);
    Bp[idx] = (bf16)W[(size_t)kk * N + cc];
}

// ---------------------------------------------------------------------------
// WMMA GEMM: C[M,N] = act(A @ W + bias)
// A given as up to 4 bf16 segments, each [M,1024] row-major (segment s covers
// k in [1024*s, 1024*(s+1)) ) -> implements concat() for free.
// Bp = packed fragments of W[K,N].  Optional fp32 and bf16 outputs.
// Block = 256 threads = 8 waves; wave w owns rows [blk.x*128+16w, +16),
// cols [blk.y*128, +128) as 8 accumulator tiles.
// Out-of-range rows are CLAMPED (not zeroed): A row r only feeds C row r,
// and rows >= M are never stored, so no exec-mask games in the hot loop.
// ---------------------------------------------------------------------------
__global__ void __launch_bounds__(256)
k_wmma_gemm(const bf16* __restrict__ A0, const bf16* __restrict__ A1,
            const bf16* __restrict__ A2, const bf16* __restrict__ A3,
            const bf16* __restrict__ Bp, const float* __restrict__ bias,
            float* __restrict__ C32, long long ldc32,
            bf16*  __restrict__ C16, long long ldc16,
            int M, int N, int K, int act_tanh) {
    const int lane = threadIdx.x & 31;
    const int wave = threadIdx.x >> 5;
    const int r    = lane & 15;
    const int half = lane >> 4;
    const int rowBase = blockIdx.x * 128 + wave * 16;
    const int colBase = blockIdx.y * 128;
    const int nCt = N >> 4;

    int row = rowBase + r;
    if (row >= M) row = 0;                       // clamp: harmless garbage rows

    v8f acc[8];
#pragma unroll
    for (int t = 0; t < 8; ++t)
#pragma unroll
        for (int j = 0; j < 8; ++j) acc[t][j] = 0.0f;

    const size_t bstride = (size_t)nCt * 512;    // Bp elems per 32-deep k tile
    const bf16* BpBase = Bp + (size_t)(colBase >> 4) * 512 + lane * 16;

    const int nseg = K >> 10;
    for (int seg = 0; seg < nseg; ++seg) {
        const bf16* As;
        switch (seg) {
            case 0:  As = A0; break;
            case 1:  As = A1; break;
            case 2:  As = A2; break;
            default: As = A3; break;
        }
        const bf16* Ab = As + (size_t)row * 1024 + 8 * half;
        const bf16* Bk = BpBase + (size_t)seg * 32 * bstride;
#pragma unroll 1
        for (int kk = 0; kk < 1024; kk += 32) {
            v8bf lo = *(const v8bf*)(Ab + kk);        // K = k0 + 8*half + 0..7
            v8bf hi = *(const v8bf*)(Ab + kk + 16);   // K = k0 + 16 + 8*half + 0..7
            v16bf a = __builtin_shufflevector(lo, hi, 0, 1, 2, 3, 4, 5, 6, 7,
                                                      8, 9, 10, 11, 12, 13, 14, 15);
#pragma unroll
            for (int t = 0; t < 8; ++t) {
                v16bf b = *(const v16bf*)(Bk + (size_t)t * 512);
                acc[t] = __builtin_amdgcn_wmma_f32_16x16x32_bf16(
                    false, a, false, b, (short)0, acc[t], false, false);
            }
            Bk += bstride;
        }
    }

    // ---- epilogue: bias + optional tanh, dual-precision store ----
#pragma unroll
    for (int t = 0; t < 8; ++t) {
        const int c = colBase + 16 * t + r;
        const float bv = bias ? bias[c] : 0.0f;
#pragma unroll
        for (int j = 0; j < 8; ++j) {
            const int m = rowBase + 8 * half + j;
            if (m < M) {
                float v = acc[t][j] + bv;
                if (act_tanh) v = tanhf(v);
                if (C32) C32[(size_t)m * ldc32 + c] = v;
                if (C16) C16[(size_t)m * ldc16 + c] = (bf16)v;
            }
        }
    }
}

// ---------------------------------------------------------------------------
// Attention scores: scores[path,b,n] = (q[path,b,:] . K[b,n,:]) / 32
// path 0: K = vice_k (queried by q_form); path 1: K = form_k (q_vice).
// grid (16 n-chunks, 8 b, 2 path), 256 threads; wave handles 32 n values.
// ---------------------------------------------------------------------------
__global__ void __launch_bounds__(256)
k_attn_scores(const float* __restrict__ q, const bf16* __restrict__ vice_k,
              const bf16* __restrict__ form_k, float* __restrict__ scores) {
    const int chunk = blockIdx.x, b = blockIdx.y, path = blockIdx.z;
    const bf16* Kt = (path == 0) ? vice_k : form_k;
    __shared__ float qs[1024];
    for (int i = threadIdx.x; i < 1024; i += 256)
        qs[i] = q[((size_t)path * 8 + b) * 1024 + i];
    __syncthreads();
    const int wave = threadIdx.x >> 5, lane = threadIdx.x & 31;
    for (int t = 0; t < 32; ++t) {
        const int n = chunk * 256 + wave * 32 + t;
        const bf16* Kr = Kt + ((size_t)b * 4096 + n) * 1024;
        float acc = 0.0f;
#pragma unroll 8
        for (int j = 0; j < 32; ++j)
            acc += (float)Kr[j * 32 + lane] * qs[j * 32 + lane];
        for (int off = 16; off; off >>= 1) acc += __shfl_xor(acc, off, 32);
        if (lane == 0)
            scores[((size_t)path * 8 + b) * 4096 + n] = acc * 0.03125f;
    }
}

// Softmax over 4096 per (path,b) row, in place.
__global__ void __launch_bounds__(256) k_softmax4096(float* __restrict__ s) {
    float* row = s + (size_t)blockIdx.x * 4096;
    __shared__ float red[256];
    const int tid = threadIdx.x;
    float m = -1e30f;
    for (int i = tid; i < 4096; i += 256) m = fmaxf(m, row[i]);
    red[tid] = m; __syncthreads();
    for (int off = 128; off; off >>= 1) {
        if (tid < off) red[tid] = fmaxf(red[tid], red[tid + off]);
        __syncthreads();
    }
    m = red[0]; __syncthreads();
    float sum = 0.0f;
    for (int i = tid; i < 4096; i += 256) {
        float e = __expf(row[i] - m);
        row[i] = e; sum += e;
    }
    red[tid] = sum; __syncthreads();
    for (int off = 128; off; off >>= 1) {
        if (tid < off) red[tid] += red[tid + off];
        __syncthreads();
    }
    const float inv = 1.0f / red[0];
    __syncthreads();
    for (int i = tid; i < 4096; i += 256) row[i] *= inv;
}

// a[path,b,h] = sum_n probs[path,b,n] * V[b,n,h]; V column reads coalesced.
__global__ void __launch_bounds__(256)
k_attn_av(const float* __restrict__ scores, const bf16* __restrict__ vice_v,
          const bf16* __restrict__ form_v, bf16* __restrict__ a_out) {
    const int hc = blockIdx.x, b = blockIdx.y, path = blockIdx.z;
    const bf16* V = (path == 0) ? vice_v : form_v;
    const float* p = scores + ((size_t)path * 8 + b) * 4096;
    const int h = hc * 256 + threadIdx.x;
    const bf16* Vb = V + (size_t)b * 4096 * 1024 + h;
    __shared__ float pl[512];
    float acc = 0.0f;
    for (int n0 = 0; n0 < 4096; n0 += 512) {
        for (int i = threadIdx.x; i < 512; i += 256) pl[i] = p[n0 + i];
        __syncthreads();
        for (int n = 0; n < 512; ++n)
            acc += pl[n] * (float)Vb[(size_t)(n0 + n) * 1024];
        __syncthreads();
    }
    a_out[((size_t)path * 8 + b) * 1024 + h] = (bf16)acc;
}

// Tail layer 2: prob[b] = sigmoid(tailhid[b,:] . W2 + b2); wave b, lane-split.
__global__ void __launch_bounds__(256)
k_tail2(const bf16* __restrict__ th, const float* __restrict__ W2,
        const float* __restrict__ b2, float* __restrict__ probs, int step) {
    const int b = threadIdx.x >> 5, lane = threadIdx.x & 31;
    float acc = 0.0f;
#pragma unroll 8
    for (int j = 0; j < 128; ++j)
        acc += (float)th[(size_t)b * 4096 + j * 32 + lane] * W2[j * 32 + lane];
    for (int off = 16; off; off >>= 1) acc += __shfl_xor(acc, off, 32);
    if (lane == 0) {
        float z = acc + b2[0];
        probs[b * 16 + step] = 1.0f / (1.0f + __expf(-z));
    }
}

// loss = -mean over (b,s) of goal*log(p) + (1-goal)*log(1-p)
__global__ void __launch_bounds__(128)
k_loss(const float* __restrict__ probs, const int* __restrict__ goal,
       float* __restrict__ out) {
    const int tid = threadIdx.x;
    const int b = tid >> 4, s = tid & 15;
    float p = probs[b * 16 + s];
    p = fminf(fmaxf(p, 1e-7f), 1.0f - 1e-7f);
    const float g = (s >= goal[b]) ? 1.0f : 0.0f;
    const float term = g * __logf(p) + (1.0f - g) * __logf(1.0f - p);
    __shared__ float red[128];
    red[tid] = term; __syncthreads();
    for (int off = 64; off; off >>= 1) {
        if (tid < off) red[tid] += red[tid + off];
        __syncthreads();
    }
    if (tid == 0) out[0] = -red[0] / 128.0f;
}

// ---------------------------------------------------------------------------
// Host side
// ---------------------------------------------------------------------------
static inline void gemm(hipStream_t st, const bf16* A0, const bf16* A1,
                        const bf16* A2, const bf16* A3, const bf16* Bp,
                        const float* bias, float* C32, long long ldc32,
                        bf16* C16, long long ldc16, int M, int N, int K,
                        int act) {
    dim3 grid((M + 127) / 128, N / 128);
    k_wmma_gemm<<<grid, 256, 0, st>>>(A0, A1, A2, A3, Bp, bias, C32, ldc32,
                                      C16, ldc16, M, N, K, act);
}

static inline void packw(hipStream_t st, const float* W, bf16* Bp, int K, int N) {
    size_t total = (size_t)K * N;
    k_pack_w<<<(unsigned)((total + 255) / 256), 256, 0, st>>>(W, Bp, K, N);
}

extern "C" void kernel_launch(void* const* d_in, const int* in_sizes, int n_in,
                              void* d_out, int out_size, void* d_ws, size_t ws_size,
                              hipStream_t stream) {
    (void)in_sizes; (void)n_in; (void)out_size; (void)ws_size;
    constexpr int H = 1024, PEAK = 16;
    constexpr size_t BN = 8ull * 4096ull;              // 32768 rows

    const float* form = (const float*)d_in[0];
    const float* vice = (const float*)d_in[1];
    const int*   goal = (const int*)d_in[2];
    // params (flattened dict order)
    const float* fk_W1 = (const float*)d_in[4];  const float* fk_b1 = (const float*)d_in[5];
    const float* fk_W2 = (const float*)d_in[6];  const float* fk_b2 = (const float*)d_in[7];
    const float* fv_W1 = (const float*)d_in[8];  const float* fv_b1 = (const float*)d_in[9];
    const float* fv_W2 = (const float*)d_in[10]; const float* fv_b2 = (const float*)d_in[11];
    const float* vk_W1 = (const float*)d_in[12]; const float* vk_b1 = (const float*)d_in[13];
    const float* vk_W2 = (const float*)d_in[14]; const float* vk_b2 = (const float*)d_in[15];
    const float* vv_W1 = (const float*)d_in[16]; const float* vv_b1 = (const float*)d_in[17];
    const float* vv_W2 = (const float*)d_in[18]; const float* vv_b2 = (const float*)d_in[19];
    const float* qf_W  = (const float*)d_in[20]; const float* qf_b  = (const float*)d_in[21];
    const float* qv_W  = (const float*)d_in[22]; const float* qv_b  = (const float*)d_in[23];
    const float* of_W1 = (const float*)d_in[24]; const float* of_b1 = (const float*)d_in[25];
    const float* of_W2 = (const float*)d_in[26]; const float* of_b2 = (const float*)d_in[27];
    const float* ov_W1 = (const float*)d_in[28]; const float* ov_b1 = (const float*)d_in[29];
    const float* ov_W2 = (const float*)d_in[30]; const float* ov_b2 = (const float*)d_in[31];
    const float* tl_W1 = (const float*)d_in[32]; const float* tl_b1 = (const float*)d_in[33];
    const float* tl_W2 = (const float*)d_in[34]; const float* tl_b2 = (const float*)d_in[35];

    float* out = (float*)d_out;                         // [form 131072][vice 131072][loss 1]

    // -------- workspace layout (bump allocator, 256B aligned) --------
    char* wc = (char*)d_ws;
    auto alloc = [&](size_t bytes) -> void* {
        void* r = (void*)wc; wc += (bytes + 255) & ~(size_t)255; return r;
    };
    bf16* formB  = (bf16*)alloc(BN * H * 2);
    bf16* viceB  = (bf16*)alloc(BN * H * 2);
    bf16* hidden = (bf16*)alloc(BN * H * 2);
    bf16* fkO = (bf16*)alloc(BN * H * 2);
    bf16* fvO = (bf16*)alloc(BN * H * 2);
    bf16* vkO = (bf16*)alloc(BN * H * 2);
    bf16* vvO = (bf16*)alloc(BN * H * 2);
    bf16* fkW1p = (bf16*)alloc(2048 * 1024 * 2); bf16* fkW2p = (bf16*)alloc(1024 * 1024 * 2);
    bf16* fvW1p = (bf16*)alloc(2048 * 1024 * 2); bf16* fvW2p = (bf16*)alloc(1024 * 1024 * 2);
    bf16* vkW1p = (bf16*)alloc(2048 * 1024 * 2); bf16* vkW2p = (bf16*)alloc(1024 * 1024 * 2);
    bf16* vvW1p = (bf16*)alloc(2048 * 1024 * 2); bf16* vvW2p = (bf16*)alloc(1024 * 1024 * 2);
    bf16* qfWp  = (bf16*)alloc(1024 * 1024 * 2); bf16* qvWp  = (bf16*)alloc(1024 * 1024 * 2);
    bf16* ofW1p = (bf16*)alloc(2048 * 1024 * 2); bf16* ofW2p = (bf16*)alloc(1024 * 1024 * 2);
    bf16* ovW1p = (bf16*)alloc(2048 * 1024 * 2); bf16* ovW2p = (bf16*)alloc(1024 * 1024 * 2);
    bf16* tlW1p = (bf16*)alloc((size_t)4096 * 4096 * 2);
    bf16* sF[2] = {(bf16*)alloc(8 * 1024 * 2), (bf16*)alloc(8 * 1024 * 2)};
    bf16* sV[2] = {(bf16*)alloc(8 * 1024 * 2), (bf16*)alloc(8 * 1024 * 2)};
    float* qbuf    = (float*)alloc(2 * 8 * 1024 * 4);   // [path][b][h]
    float* scoresB = (float*)alloc(2 * 8 * 4096 * 4);   // [path][b][n]
    bf16*  abuf    = (bf16*)alloc(2 * 8 * 1024 * 2);    // [path][b][h]
    bf16*  hidsF   = (bf16*)alloc(8 * 1024 * 2);
    bf16*  hidsV   = (bf16*)alloc(8 * 1024 * 2);
    bf16*  tailhid = (bf16*)alloc(8 * 4096 * 2);
    float* probs   = (float*)alloc(128 * 4);            // [b][step]

    // -------- one-time converts / packs --------
    const size_t nbig = BN * H;
    k_f2bf<<<(unsigned)((nbig + 255) / 256), 256, 0, stream>>>(form, formB, nbig);
    k_f2bf<<<(unsigned)((nbig + 255) / 256), 256, 0, stream>>>(vice, viceB, nbig);
    packw(stream, fk_W1, fkW1p, 2048, 1024); packw(stream, fk_W2, fkW2p, 1024, 1024);
    packw(stream, fv_W1, fvW1p, 2048, 1024); packw(stream, fv_W2, fvW2p, 1024, 1024);
    packw(stream, vk_W1, vkW1p, 2048, 1024); packw(stream, vk_W2, vkW2p, 1024, 1024);
    packw(stream, vv_W1, vvW1p, 2048, 1024); packw(stream, vv_W2, vvW2p, 1024, 1024);
    packw(stream, qf_W, qfWp, 1024, 1024);   packw(stream, qv_W, qvWp, 1024, 1024);
    packw(stream, of_W1, ofW1p, 2048, 1024); packw(stream, of_W2, ofW2p, 1024, 1024);
    packw(stream, ov_W1, ovW1p, 2048, 1024); packw(stream, ov_W2, ovW2p, 1024, 1024);
    packw(stream, tl_W1, tlW1p, 4096, 4096);
    k_zero_bf16<<<32, 256, 0, stream>>>(sF[0], 8192);
    k_zero_bf16<<<32, 256, 0, stream>>>(sV[0], 8192);

    // -------- hoisted K/V projection MLPs (the heavy GEMMs) --------
    struct { bf16* W1p; const float* b1; bf16* W2p; const float* b2; bf16* O; } mlps[4] = {
        {fkW1p, fk_b1, fkW2p, fk_b2, fkO}, {fvW1p, fv_b1, fvW2p, fv_b2, fvO},
        {vkW1p, vk_b1, vkW2p, vk_b2, vkO}, {vvW1p, vv_b1, vvW2p, vv_b2, vvO}};
    for (int m = 0; m < 4; ++m) {
        gemm(stream, formB, viceB, nullptr, nullptr, mlps[m].W1p, mlps[m].b1,
             nullptr, 0, hidden, 1024, (int)BN, 1024, 2048, /*tanh*/1);
        gemm(stream, hidden, nullptr, nullptr, nullptr, mlps[m].W2p, mlps[m].b2,
             nullptr, 0, mlps[m].O, 1024, (int)BN, 1024, 1024, /*tanh*/0);
    }

    // -------- halting loop (16 sequential steps) --------
    for (int s = 0; s < PEAK; ++s) {
        const int cur = s & 1, nxt = cur ^ 1;
        // q projections (M=8 rows, clamped rows inside the WMMA tile)
        gemm(stream, sF[cur], nullptr, nullptr, nullptr, qfWp, qf_b,
             qbuf, 1024, nullptr, 0, 8, 1024, 1024, 0);
        gemm(stream, sV[cur], nullptr, nullptr, nullptr, qvWp, qv_b,
             qbuf + 8 * 1024, 1024, nullptr, 0, 8, 1024, 1024, 0);
        // attention (path0: q_form vs vice_k/vice_v; path1: q_vice vs form_k/form_v)
        k_attn_scores<<<dim3(16, 8, 2), 256, 0, stream>>>(qbuf, vkO, fkO, scoresB);
        k_softmax4096<<<16, 256, 0, stream>>>(scoresB);
        k_attn_av<<<dim3(4, 8, 2), 256, 0, stream>>>(scoresB, vvO, fvO, abuf);
        // o_form: concat(old_form, a_form) -> tanh -> linear; outputs fp32 + next state
        gemm(stream, sF[cur], abuf, nullptr, nullptr, ofW1p, of_b1,
             nullptr, 0, hidsF, 1024, 8, 1024, 2048, 1);
        gemm(stream, hidsF, nullptr, nullptr, nullptr, ofW2p, of_b2,
             out + (size_t)s * 1024, 16384, sF[nxt], 1024, 8, 1024, 1024, 0);
        // o_vice
        gemm(stream, sV[cur], abuf + 8 * 1024, nullptr, nullptr, ovW1p, ov_b1,
             nullptr, 0, hidsV, 1024, 8, 1024, 2048, 1);
        gemm(stream, hidsV, nullptr, nullptr, nullptr, ovW2p, ov_b2,
             out + 131072 + (size_t)s * 1024, 16384, sV[nxt], 1024, 8, 1024, 1024, 0);
        // halting head: concat(old_f, old_v, new_f, new_v) [8,4096] -> tanh -> dot
        gemm(stream, sF[cur], sV[cur], sF[nxt], sV[nxt], tlW1p, tl_b1,
             nullptr, 0, tailhid, 4096, 8, 4096, 4096, 1);
        k_tail2<<<1, 256, 0, stream>>>(tailhid, tl_W2, tl_b2, probs, s);
    }

    k_loss<<<1, 128, 0, stream>>>(probs, goal, out + 262144);
}
